// TransformerBlockQuantum_65481071396062
// MI455X (gfx1250) — matrix operand order repaired and checked by reading, not compiled, not run on hardware
//
#include <hip/hip_runtime.h>
#include <hip/hip_bf16.h>

typedef __attribute__((ext_vector_type(16))) _Float16 v16h;
typedef __attribute__((ext_vector_type(8)))  float    v8f;
typedef __attribute__((ext_vector_type(8)))  unsigned int v8u;

#define SWZ_SWAPX16 0x401f   // ds_swizzle: group-of-32, xor_mask=0x10, and_mask=0x1f

// ---------- helpers ----------
__device__ __forceinline__ v16h load_v16h_2x(const _Float16* p0, const _Float16* p1) {
  uint4 a = *(const uint4*)p0;
  uint4 b = *(const uint4*)p1;
  v8u u;
  u[0] = a.x; u[1] = a.y; u[2] = a.z; u[3] = a.w;
  u[4] = b.x; u[5] = b.y; u[6] = b.z; u[7] = b.w;
  return __builtin_bit_cast(v16h, u);
}

__device__ __forceinline__ unsigned pack_f16x2(float a, float b) {
  _Float16 ha = (_Float16)a, hb = (_Float16)b;
  unsigned short ua = __builtin_bit_cast(unsigned short, ha);
  unsigned short ub = __builtin_bit_cast(unsigned short, hb);
  return (unsigned)ua | ((unsigned)ub << 16);
}

__device__ __forceinline__ float swz16f(float v) {
  int i = __builtin_bit_cast(int, v);
  i = __builtin_amdgcn_ds_swizzle(i, SWZ_SWAPX16);
  return __builtin_bit_cast(float, i);
}

// ---------- workspace layout (bytes) ----------
//  mh  : 5120*32 f16  (cos(x1+thetaF), cols 10..31 zero)         @ 0       (327680 B)
//  w1h : 4096*32 f16  (w1 rows, K padded to 32 with zeros)       @ 327680  (262144 B)
//  w2h : 16*4096 f16  (w2 rows, rows 10..15 zero)                @ 589824  (131072 B)
//  x1p : 5120*16 f32  (x1 rows, cols 10..15 zero)                @ 720896  (327680 B)
// total = 1048576 B

#define W1H_N  131072
#define W2H_N   65536
#define MH_N   163840
#define X1P_N   81920

__global__ __launch_bounds__(256) void prep_kernel(
    const float* __restrict__ w1, const float* __restrict__ w2,
    _Float16* __restrict__ w1h, _Float16* __restrict__ w2h,
    _Float16* __restrict__ mh, float* __restrict__ x1p)
{
  int total = W1H_N + W2H_N + MH_N + X1P_N;
  for (int idx = blockIdx.x * blockDim.x + threadIdx.x; idx < total;
       idx += gridDim.x * blockDim.x) {
    int i = idx;
    if (i < W1H_N) {
      int n = i >> 5, k = i & 31;
      w1h[i] = (k < 10) ? (_Float16)w1[n * 10 + k] : (_Float16)0.0f;
    } else if ((i -= W1H_N) < W2H_N) {
      int e = i >> 12;                       // w2h[e][c], linear index matches w2
      w2h[i] = (e < 10) ? (_Float16)w2[i] : (_Float16)0.0f;
    } else if ((i -= W2H_N) < MH_N) {
      mh[i] = (_Float16)0.0f;
    } else {
      i -= MH_N;
      x1p[i] = 0.0f;
    }
  }
}

// ---------- attention + LN1 + FFN quantum encode (per batch) ----------
__global__ __launch_bounds__(128) void attn_kernel(
    const float* __restrict__ x, const float* __restrict__ attn_theta,
    const float* __restrict__ ffn_theta, const float* __restrict__ g1,
    const float* __restrict__ beta1,
    float* __restrict__ x1p, _Float16* __restrict__ mh)
{
  __shared__ float xs[10][10];
  __shared__ float qk[10][10];
  __shared__ float sc[10][10];   // scores, then softmax weights (in place)
  __shared__ float r1[10][10];   // x + attn_out
  __shared__ float th[10], fth[10], g1s[10], b1s[10], mu[10], rs[10];

  const int t = threadIdx.x;
  const int b = blockIdx.x;
  const float* xb = x + b * 100;

  if (t < 100) xs[t / 10][t % 10] = xb[t];
  if (t < 10) {
    th[t]  = attn_theta[t];
    fth[t] = ffn_theta[t];
    g1s[t] = g1[t];
    b1s[t] = beta1[t];
  }
  __syncthreads();

  // qkv[s][i] = prod_{j<=i} cos(x[s][j] + thetaA[j])   (CNOT-ladder collapse)
  if (t < 10) {
    float p = 1.0f;
    #pragma unroll
    for (int j = 0; j < 10; ++j) {
      p *= cosf(xs[t][j] + th[j]);
      qk[t][j] = p;
    }
  }
  __syncthreads();

  // scores[i][j] = sum_s qkv[s][i]*qkv[s][j] / sqrt(10)
  if (t < 100) {
    int i = t / 10, j = t % 10;
    float a = 0.0f;
    #pragma unroll
    for (int s = 0; s < 10; ++s) a += qk[s][i] * qk[s][j];
    sc[i][j] = a * 0.31622776601683794f;
  }
  __syncthreads();

  // row softmax (in place)
  if (t < 10) {
    float mx = sc[t][0];
    #pragma unroll
    for (int j = 1; j < 10; ++j) mx = fmaxf(mx, sc[t][j]);
    float sum = 0.0f;
    #pragma unroll
    for (int j = 0; j < 10; ++j) { float e = expf(sc[t][j] - mx); sc[t][j] = e; sum += e; }
    float inv = 1.0f / sum;
    #pragma unroll
    for (int j = 0; j < 10; ++j) sc[t][j] *= inv;
  }
  __syncthreads();

  // attn_out + residual
  if (t < 100) {
    int i = t / 10, k = t % 10;
    float a = 0.0f;
    #pragma unroll
    for (int j = 0; j < 10; ++j) a += sc[i][j] * qk[j][k];
    r1[i][k] = xs[i][k] + a;
  }
  __syncthreads();

  // LayerNorm 1 stats
  if (t < 10) {
    float m = 0.0f;
    #pragma unroll
    for (int e = 0; e < 10; ++e) m += r1[t][e];
    m *= 0.1f;
    float v = 0.0f;
    #pragma unroll
    for (int e = 0; e < 10; ++e) { float d = r1[t][e] - m; v += d * d; }
    v *= 0.1f;
    mu[t] = m;
    rs[t] = rsqrtf(v + 1e-5f);
  }
  __syncthreads();

  // write x1 (padded) and m = cos(x1 + thetaF) as f16 (padded buffer pre-zeroed)
  if (t < 100) {
    int s = t / 10, e = t % 10;
    float v = (r1[s][e] - mu[s]) * rs[s] * g1s[e] + b1s[e];
    int row = b * 10 + s;
    x1p[row * 16 + e] = v;
    mh[row * 32 + e]  = (_Float16)cosf(v + fth[e]);
  }
}

// ---------- fused FFN: h = relu(m w1^T + b1); out = LN2(x1 + h w2^T + b2) ----------
// One wave per 16-row tile; everything register-resident between the two GEMMs.
__global__ __launch_bounds__(128) void ffn_kernel(
    const float* __restrict__ b1, const float* __restrict__ b2,
    const float* __restrict__ g2, const float* __restrict__ beta2,
    const _Float16* __restrict__ mh, const _Float16* __restrict__ w1h,
    const _Float16* __restrict__ w2h, const float* __restrict__ x1p,
    float* __restrict__ out)
{
  const int lane = threadIdx.x & 31;
  const int wave = threadIdx.x >> 5;
  const int tile = blockIdx.x * 4 + wave;       // 0..319
  const int lm   = lane & 15;
  const int hi   = lane >> 4;                   // 0 or 1
  const int hi8  = hi * 8;
  const int row  = tile * 16 + lm;              // global token row 0..5119

  // B operand of GEMM1: m^T (K=embed padded 32, N=row). Lane holds N=lm, K = hi8..hi8+7, 16+hi8..16+hi8+7
  const _Float16* mrow = mh + row * 32 + hi8;
  const v16h Bm = load_v16h_2x(mrow, mrow + 16);

  v8f acc = {};   // out^T accumulator: M=embed dim, N=row

  for (int n0 = 0; n0 < 4096; n0 += 32) {
    // GEMM1: h^T chunks = w1_chunk (16x32) x m^T (32x16)
    const _Float16* a1a = w1h + (n0 + lm) * 32 + hi8;
    const _Float16* a1b = w1h + (n0 + 16 + lm) * 32 + hi8;
    v16h A1a = load_v16h_2x(a1a, a1a + 16);
    v16h A1b = load_v16h_2x(a1b, a1b + 16);
    v8f z = {};
    v8f ht0 = __builtin_amdgcn_wmma_f32_16x16x32_f16(false, A1a, false, Bm, (short)0, z, false, false);
    v8f ht1 = __builtin_amdgcn_wmma_f32_16x16x32_f16(false, A1b, false, Bm, (short)0, z, false, false);

    // bias + ReLU in h^T layout: lane's ffn dims are n0 + hi8 + r (ht0), +16 (ht1)
    const float4* bv0 = (const float4*)(b1 + n0 + hi8);
    const float4* bv1 = (const float4*)(b1 + n0 + 16 + hi8);
    float4 ba = bv0[0], bb = bv0[1], bc = bv1[0], bd = bv1[1];
    float bb0[8] = {ba.x, ba.y, ba.z, ba.w, bb.x, bb.y, bb.z, bb.w};
    float bb1[8] = {bc.x, bc.y, bc.z, bc.w, bd.x, bd.y, bd.z, bd.w};
    float h0[8], h1[8];
    #pragma unroll
    for (int r = 0; r < 8; ++r) {
      h0[r] = fmaxf(ht0[r] + bb0[r], 0.0f);
      h1[r] = fmaxf(ht1[r] + bb1[r], 0.0f);
    }

    // Repack h^T (C layout) -> B operand layout for GEMM2.
    // Lane L and L^16 share the same row; swap the K-half-blocks via ds_swizzle.
    unsigned u0[4], u1[4], o0[4], o1[4];
    #pragma unroll
    for (int r = 0; r < 4; ++r) {
      u0[r] = pack_f16x2(h0[2 * r], h0[2 * r + 1]);
      u1[r] = pack_f16x2(h1[2 * r], h1[2 * r + 1]);
    }
    #pragma unroll
    for (int r = 0; r < 4; ++r) {
      o0[r] = (unsigned)__builtin_amdgcn_ds_swizzle((int)u0[r], SWZ_SWAPX16);
      o1[r] = (unsigned)__builtin_amdgcn_ds_swizzle((int)u1[r], SWZ_SWAPX16);
    }
    v8u bu;
    #pragma unroll
    for (int r = 0; r < 4; ++r) {
      bu[r]     = hi ? o1[r] : u0[r];   // K = 0..7   (lo half)  or 16..23
      bu[4 + r] = hi ? u1[r] : o0[r];   // K = 8..15  (lo half)  or 24..31
    }
    v16h B2 = __builtin_bit_cast(v16h, bu);

    // GEMM2: out^T += w2_chunk (16x32) x h^T_chunk (32x16)
    const _Float16* a2 = w2h + lm * 4096 + n0 + hi * 16;
    v16h A2 = load_v16h_2x(a2, a2 + 8);
    acc = __builtin_amdgcn_wmma_f32_16x16x32_f16(false, A2, false, B2, (short)0, acc, false, false);
  }

  // Epilogue: bias b2, residual x1, LayerNorm2 over the 10 embed dims.
  // Lane holds embed dims m = hi8 + r for token `row`; valid m < 10.
  float bb2[8] = {0,0,0,0,0,0,0,0};
  float gg[8]  = {0,0,0,0,0,0,0,0};
  float be[8]  = {0,0,0,0,0,0,0,0};
  if (hi == 0) {
    float4 p0 = *(const float4*)(b2);     float4 p1 = *(const float4*)(b2 + 4);
    float4 q0 = *(const float4*)(g2);     float4 q1 = *(const float4*)(g2 + 4);
    float4 r0 = *(const float4*)(beta2);  float4 r1v = *(const float4*)(beta2 + 4);
    bb2[0]=p0.x; bb2[1]=p0.y; bb2[2]=p0.z; bb2[3]=p0.w; bb2[4]=p1.x; bb2[5]=p1.y; bb2[6]=p1.z; bb2[7]=p1.w;
    gg[0]=q0.x;  gg[1]=q0.y;  gg[2]=q0.z;  gg[3]=q0.w;  gg[4]=q1.x;  gg[5]=q1.y;  gg[6]=q1.z;  gg[7]=q1.w;
    be[0]=r0.x;  be[1]=r0.y;  be[2]=r0.z;  be[3]=r0.w;  be[4]=r1v.x; be[5]=r1v.y; be[6]=r1v.z; be[7]=r1v.w;
  } else {
    bb2[0] = b2[8];    bb2[1] = b2[9];
    gg[0]  = g2[8];    gg[1]  = g2[9];
    be[0]  = beta2[8]; be[1]  = beta2[9];
  }

  const float4* xr = (const float4*)(x1p + row * 16 + hi8);
  float4 x0 = xr[0], x1v = xr[1];
  float xv[8] = {x0.x, x0.y, x0.z, x0.w, x1v.x, x1v.y, x1v.z, x1v.w};

  float y[8];
  float s1 = 0.0f, s2 = 0.0f;
  #pragma unroll
  for (int r = 0; r < 8; ++r) {
    float v = xv[r] + acc[r] + bb2[r];
    y[r] = v;
    if (hi == 0 || r < 2) { s1 += v; s2 += v * v; }   // only m<10 contribute
  }
  float tot1 = s1 + swz16f(s1);   // combine lane L with lane L^16 (same token row)
  float tot2 = s2 + swz16f(s2);
  float mean = tot1 * 0.1f;
  float var  = tot2 * 0.1f - mean * mean;
  float rstd = rsqrtf(var + 1e-5f);

  float o[8];
  #pragma unroll
  for (int r = 0; r < 8; ++r) o[r] = (y[r] - mean) * rstd * gg[r] + be[r];

  float* orow = out + row * 10;
  if (hi == 0) {
    #pragma unroll
    for (int r = 0; r < 4; ++r) {
      float2 tv; tv.x = o[2 * r]; tv.y = o[2 * r + 1];
      *(float2*)(orow + 2 * r) = tv;
    }
  } else {
    float2 tv; tv.x = o[0]; tv.y = o[1];
    *(float2*)(orow + 8) = tv;
  }
}

extern "C" void kernel_launch(void* const* d_in, const int* in_sizes, int n_in,
                              void* d_out, int out_size, void* d_ws, size_t ws_size,
                              hipStream_t stream) {
  const float* x          = (const float*)d_in[0];
  const float* attn_theta = (const float*)d_in[1];
  const float* ffn_theta  = (const float*)d_in[2];
  const float* w1         = (const float*)d_in[3];
  const float* b1         = (const float*)d_in[4];
  const float* w2         = (const float*)d_in[5];
  const float* b2         = (const float*)d_in[6];
  const float* g1         = (const float*)d_in[7];
  const float* beta1      = (const float*)d_in[8];
  const float* g2         = (const float*)d_in[9];
  const float* beta2      = (const float*)d_in[10];
  float* out = (float*)d_out;

  char* ws = (char*)d_ws;
  _Float16* mh  = (_Float16*)(ws);             // 5120*32 f16
  _Float16* w1h = (_Float16*)(ws + 327680);    // 4096*32 f16
  _Float16* w2h = (_Float16*)(ws + 589824);    // 16*4096 f16
  float*    x1p = (float*)   (ws + 720896);    // 5120*16 f32

  prep_kernel<<<256, 256, 0, stream>>>(w1, w2, w1h, w2h, mh, x1p);
  attn_kernel<<<512, 128, 0, stream>>>(x, attn_theta, ffn_theta, g1, beta1, x1p, mh);
  ffn_kernel<<<80, 128, 0, stream>>>(b1, b2, g2, beta2, mh, w1h, w2h, x1p, out);
}
